// EdgeNetwork_66254165508977
// MI455X (gfx1250) — matrix operand order repaired and checked by reading, not compiled, not run on hardware
//
#include <hip/hip_runtime.h>

typedef _Float16 v16h __attribute__((ext_vector_type(16)));
typedef __fp16   v2fp __attribute__((ext_vector_type(2)));   // cvt_pkrtz return type
typedef float    v8f  __attribute__((ext_vector_type(8)));

#define NCOL    1024          // D*D
#define WPB     8             // waves per block
#define THREADS (WPB * 32)
#define WPITCH  12            // dwords per packed-W row: [kp0..3 | bias,0,0,0 | kp4..7]
#define WTOTAL  (NCOL * WPITCH + 4)   // +4 zero tail dwords (last row, h=1 upper half)

union PK { v2fp h; unsigned int u; };
union AB { unsigned int u[8]; uint4 q[2]; v16h v; };

__device__ __forceinline__ unsigned int pk_f16(float a, float b) {
  PK p; p.h = __builtin_amdgcn_cvt_pkrtz(a, b); return p.u;
}

__device__ __forceinline__ float relu1(float x) {
  // single v_med3_num_f32: clamp to [0, 3e38] -- finite bound so InstCombine
  // cannot fold it back into (canonicalize + maxnum)
  return __builtin_amdgcn_fmed3f(x, 0.0f, 3.0e38f);
}

__global__ __launch_bounds__(THREADS)
void edge_net_wmma(const float* __restrict__ states,
                   const float* __restrict__ edges,
                   const float* __restrict__ W,
                   const float* __restrict__ bias,
                   float* __restrict__ out,
                   int numTiles)
{
  __shared__ unsigned int Wlds[WTOTAL];   // ~48 KB: Wᵀ f16-packed, pitch 12, bias inline

  const int tid = threadIdx.x;

  // ---- Prologue: stage Wᵀ (f16 k-pairs) + inline bias + zeros into LDS ----
  for (int idx = tid; idx < NCOL * 8; idx += THREADS) {
    const int kp = idx >> 10;            // k-pair 0..7
    const int n  = idx & (NCOL - 1);     // output column 0..1023 (coalesced)
    const int d  = (kp < 4) ? kp : kp + 4;          // dwords 0..3 / 8..11
    Wlds[n * WPITCH + d] = pk_f16(W[(2 * kp) * NCOL + n],
                                  W[(2 * kp + 1) * NCOL + n]);
  }
  for (int idx = tid; idx < NCOL; idx += THREADS) {
    // dwords 4..7: bias at K=16 slot (h=0 upper quad), zeros for K=18..23
    uint4 bq; bq.x = pk_f16(bias[idx], 0.f); bq.y = 0u; bq.z = 0u; bq.w = 0u;
    *reinterpret_cast<uint4*>(&Wlds[idx * WPITCH + 4]) = bq;
  }
  if (tid < 4) Wlds[NCOL * WPITCH + tid] = 0u;      // zero tail for last row
  __syncthreads();

  const int wave = tid >> 5, lane = tid & 31;
  const int e  = lane & 15;              // edge-in-tile (WMMA N / C-layout column)
  const int hh = lane >> 4;              // lane half
  const int gw = blockIdx.x * WPB + wave;
  const int nw = gridDim.x * WPB;

  for (int tile = gw; tile < numTiles; tile += nw) {
    const long rowBase = (long)tile * 16;
    float* orow = out + (rowBase + e) * 32;         // this lane's output row

    // ---- B operand (constant per tile): edges tile, f16, K padded 16->32.
    //      regs r<4: K=8*hh+2r,+1 (edge features, table order); reg 4 of
    //      lanes 0-15: K=16 = 1.0 (activates the bias slot of A). All other
    //      K rows of B are zero, which also neutralizes the junk K-slots of A. ----
    AB Bop;
    {
      const float2* er2 = reinterpret_cast<const float2*>(edges + (rowBase + e) * 16);
      #pragma unroll
      for (int r = 0; r < 8; ++r) {
        float2 w = er2[(r & 3) + 4 * hh];
        unsigned int u = pk_f16(w.x, w.y);
        Bop.u[r] = (r < 4) ? u : 0u;
      }
      Bop.u[4] = hh ? 0u : 0x00003C00u /* pack(1.0h, 0) */;
    }

    // ---- State registers: S[p][v] = states[row e][16p + v + 8*hh] ----
    float S0[8], S1[8];
    {
      const float* sr = states + (rowBase + e) * 32 + 8 * hh;
      #pragma unroll
      for (int v = 0; v < 8; ++v) { S0[v] = sr[v]; S1[v] = sr[16 + v]; }
    }

    // ---- A operands: two ds_load_b128 per half (no padding regs, bias inline).
    //      h=0 reads row dwords 0..7, h=1 reads 8..15 (tail spills into next
    //      row's data: finite f16, multiplied by zero B rows -> harmless). ----
    AB A0, A1;
    auto loadA = [&](int i) {
      const int n0 = 32 * i + e;                    // chunk 2i   row
      const int n1 = n0 + 16;                       // chunk 2i+1 row
      const unsigned int* p0 = &Wlds[n0 * WPITCH + 8 * hh];
      const unsigned int* p1 = &Wlds[n1 * WPITCH + 8 * hh];
      A0.q[0] = *reinterpret_cast<const uint4*>(p0);
      A0.q[1] = *reinterpret_cast<const uint4*>(p0 + 4);
      A1.q[0] = *reinterpret_cast<const uint4*>(p1);
      A1.q[1] = *reinterpret_cast<const uint4*>(p1 + 4);
    };

    // Consume one output element: relu * state, half-merge, direct store
    auto consume = [&](int i, const v8f& P0, const v8f& P1) {
      float acc0 = 0.f, acc1 = 0.f;                 // two chains for dual-issue
      #pragma unroll
      for (int v = 0; v < 8; ++v) {
        acc0 = fmaf(relu1(P0[v]), S0[v], acc0);
        acc1 = fmaf(relu1(P1[v]), S1[v], acc1);
      }
      float acc = acc0 + acc1;
      acc += __shfl_xor(acc, 16);                   // merge j-halves across lane halves
      if (lane < 16) orow[i] = acc;                 // row-line completed over the tile
    };

    // ---- Software-pipelined ping-pong loop: WMMAs for element k+1 issue
    //      before element k is consumed (fills the WMMA->VALU hazard window).
    //      Manual 2x unroll alternates D/E buffers -> no rotation moves, and
    //      unroll(disable) keeps the register footprint bounded. ----
    v8f D0, D1, E0, E1;
    loadA(0);
    D0 = __builtin_amdgcn_wmma_f32_16x16x32_f16(
        false, A0.v, false, Bop.v, (short)0, (v8f){}, false, false);
    D1 = __builtin_amdgcn_wmma_f32_16x16x32_f16(
        false, A1.v, false, Bop.v, (short)0, (v8f){}, false, false);

    #pragma clang loop unroll(disable)
    for (int i = 0; i < 15; ++i) {
      loadA(2 * i + 1);
      E0 = __builtin_amdgcn_wmma_f32_16x16x32_f16(
          false, A0.v, false, Bop.v, (short)0, (v8f){}, false, false);
      E1 = __builtin_amdgcn_wmma_f32_16x16x32_f16(
          false, A1.v, false, Bop.v, (short)0, (v8f){}, false, false);
      consume(2 * i, D0, D1);

      loadA(2 * i + 2);
      D0 = __builtin_amdgcn_wmma_f32_16x16x32_f16(
          false, A0.v, false, Bop.v, (short)0, (v8f){}, false, false);
      D1 = __builtin_amdgcn_wmma_f32_16x16x32_f16(
          false, A1.v, false, Bop.v, (short)0, (v8f){}, false, false);
      consume(2 * i + 1, E0, E1);
    }
    // tail: elements 30 and 31
    loadA(31);
    E0 = __builtin_amdgcn_wmma_f32_16x16x32_f16(
        false, A0.v, false, Bop.v, (short)0, (v8f){}, false, false);
    E1 = __builtin_amdgcn_wmma_f32_16x16x32_f16(
        false, A1.v, false, Bop.v, (short)0, (v8f){}, false, false);
    consume(30, D0, D1);
    consume(31, E0, E1);
  }
}

extern "C" void kernel_launch(void* const* d_in, const int* in_sizes, int n_in,
                              void* d_out, int out_size, void* d_ws, size_t ws_size,
                              hipStream_t stream) {
  const float* states = (const float*)d_in[0];   // [BS,E,32] f32
  const float* edges  = (const float*)d_in[1];   // [BS,E,16] f32
  const float* W      = (const float*)d_in[2];   // [16,1024] f32
  const float* bias   = (const float*)d_in[3];   // [1024]    f32
  float* out = (float*)d_out;                    // [BS,E,32] f32

  const int rows     = in_sizes[1] / 16;         // BS*E edge rows
  const int numTiles = rows / 16;                // 16 rows per wave-tile
  int blocks = (numTiles + WPB - 1) / WPB;
  if (blocks > 256) blocks = 256;

  hipLaunchKernelGGL(edge_net_wmma, dim3(blocks), dim3(THREADS), 0, stream,
                     states, edges, W, bias, out, numTiles);
}
